// GNN_12257836662944
// MI455X (gfx1250) — matrix operand order repaired and checked by reading, not compiled, not run on hardware
//
#include <hip/hip_runtime.h>
#include <stdint.h>

#define NF    128
#define NNODE 100000
#define NEDGE 500000
#define TPB   256

typedef __attribute__((ext_vector_type(16))) __bf16 v16bf;
typedef __attribute__((ext_vector_type(8)))  float  v8f;

union BFrag { uint32_t u[8]; v16bf v; };

__device__ __forceinline__ uint16_t f2bf(float f) {
  union { float f; uint32_t u; } v; v.f = f;
  uint32_t u = v.u;
  u += 0x7FFFu + ((u >> 16) & 1u);   // round-to-nearest-even
  return (uint16_t)(u >> 16);
}
__device__ __forceinline__ float bf2f(uint16_t b) {
  union { uint32_t u; float f; } v; v.u = ((uint32_t)b) << 16;
  return v.f;
}

// ---- CDNA5 async copy: global -> LDS, 16B per lane, tracked by ASYNCcnt -----
__device__ __forceinline__ void async_copy_b128(void* lds_dst, const void* gsrc) {
  uint32_t lds_off = (uint32_t)(uintptr_t)lds_dst;      // AS3 offset = low 32 bits
  asm volatile("global_load_async_to_lds_b128 %0, %1, off"
               :: "v"(lds_off), "v"((uint64_t)(uintptr_t)gsrc)
               : "memory");
}
__device__ __forceinline__ void async_wait0() {
  asm volatile("s_wait_asynccnt 0" ::: "memory");
}

// ---- WMMA fragment loaders (ISA 7.12.2 16-bit layouts, wave32) --------------

// A: 16xK row-major bf16 tile. lanes 0-15: row=lane, i<4 -> K=2i ; i>=4 -> K=8+2i
//    lanes 16-31: row=lane-16, i<4 -> K=8+2i ; i>=4 -> K=16+2i
__device__ __forceinline__ v16bf load_a_tile(const uint16_t* base, int lda, int kc, int lane) {
  BFrag f;
  const int row = lane & 15;
  const int hi4 = (lane & 16) ? 8 : 0;
  const uint16_t* p = base + row * lda + kc;
#pragma unroll
  for (int i = 0; i < 8; ++i) {
    int klo = 2 * i + ((i < 4) ? hi4 : hi4 + 8);
    f.u[i] = *(const uint32_t*)(p + klo);
  }
  return f.v;
}

// B: Kx16 from weights stored TRANSPOSED: Wt[n][k]
__device__ __forceinline__ v16bf load_b_w(const uint16_t* Wt, int Ktot, int kc, int n0, int lane) {
  BFrag f;
  const int col = lane & 15;
  const uint16_t* p = Wt + (size_t)(n0 + col) * Ktot + kc + ((lane & 16) ? 8 : 0);
#pragma unroll
  for (int i = 0; i < 8; ++i) f.u[i] = *(const uint32_t*)(p + 2 * i + ((i < 4) ? 0 : 8));
  return f.v;
}

__device__ __forceinline__ v8f wmma_bf16(v16bf a, v16bf b, v8f c) {
  return __builtin_amdgcn_wmma_f32_16x16x32_bf16(false, a, false, b, (short)0, c, false, false);
}

// bias + relu + bf16 store of a 16x16 D tile. dst points at row 0 of the 16-row block.
__device__ __forceinline__ void epilogue_relu_bf16(uint16_t* dst, int ldd, int n0,
                                                   v8f acc, const float* bias, int lane) {
  const int col = lane & 15;
  const float bb = bias[n0 + col];
  const int rbase = (lane & 16) ? 8 : 0;
#pragma unroll
  for (int i = 0; i < 8; ++i) {
    float x = fmaxf(acc[i] + bb, 0.f);
    dst[(size_t)(rbase + i) * ldd + n0 + col] = f2bf(x);
  }
}

// ---- weight convert+transpose+K-pad: Wt[n*Kpad+k] = W[k*NF+n] (bf16) --------
__global__ void __launch_bounds__(256) conv_w_kernel(const float* __restrict__ W,
                                                     uint16_t* __restrict__ Wt,
                                                     int Kin, int Kpad) {
  int idx = blockIdx.x * 256 + threadIdx.x;
  if (idx >= NF * Kpad) return;
  int n = idx / Kpad, k = idx % Kpad;
  float v = (k < Kin) ? W[k * NF + n] : 0.f;
  Wt[idx] = f2bf(v);
}

// ---- node encoder: pe = relu(relu([attr,state] @ w1 + b1) @ w2 + b2) --------
__global__ void __launch_bounds__(TPB) node_enc_kernel(
    const float* __restrict__ attr, const float* __restrict__ state,
    const uint16_t* __restrict__ w1t, const float* __restrict__ b1,
    const uint16_t* __restrict__ w2t, const float* __restrict__ b2,
    uint16_t* __restrict__ pe) {
  __shared__ __align__(16) uint16_t X[16 * 32];
  __shared__ __align__(16) uint16_t H[16 * NF];
  const int tid = threadIdx.x;
  const int row0 = blockIdx.x * 16;
  for (int idx = tid; idx < 16 * 32; idx += TPB) {
    int r = idx >> 5, c = idx & 31;
    int node = row0 + r;
    float v = 0.f;
    if (c < 16)      v = attr[node * 16 + c];
    else if (c < 28) v = state[node * 12 + (c - 16)];
    X[idx] = f2bf(v);
  }
  __syncthreads();
  const int lane = tid & 31, n0 = (tid >> 5) * 16;
  v8f acc = {};
  acc = wmma_bf16(load_a_tile(X, 32, 0, lane), load_b_w(w1t, 32, 0, n0, lane), acc);
  epilogue_relu_bf16(H, NF, n0, acc, b1, lane);
  __syncthreads();
  v8f acc2 = {};
#pragma unroll
  for (int kc = 0; kc < NF; kc += 32)
    acc2 = wmma_bf16(load_a_tile(H, NF, kc, lane), load_b_w(w2t, NF, kc, n0, lane), acc2);
  epilogue_relu_bf16(pe + (size_t)row0 * NF, NF, n0, acc2, b2, lane);
}

// ---- edge encoder: 3-layer MLP over gathered edge features ------------------
__global__ void __launch_bounds__(TPB) edge_enc_kernel(
    const float* __restrict__ attr, const float* __restrict__ state,
    const float* __restrict__ Ra, const int* __restrict__ recv, const int* __restrict__ send,
    const uint16_t* __restrict__ w1t, const float* __restrict__ b1,
    const uint16_t* __restrict__ w2t, const float* __restrict__ b2,
    const uint16_t* __restrict__ w3t, const float* __restrict__ b3,
    uint16_t* __restrict__ ee) {
  __shared__ __align__(16) uint16_t X[16 * 64];
  __shared__ __align__(16) uint16_t H1[16 * NF];
  __shared__ __align__(16) uint16_t H2[16 * NF];
  __shared__ int ridx[16], sidx[16];
  const int tid = threadIdx.x;
  const int e0 = blockIdx.x * 16;
  if (tid < 16)      ridx[tid] = recv[e0 + tid];
  else if (tid < 32) sidx[tid - 16] = send[e0 + tid - 16];
  __syncthreads();
  for (int idx = tid; idx < 16 * 64; idx += TPB) {
    int r = idx >> 6, c = idx & 63;
    float v = 0.f;
    if (c < 16)      v = attr[ridx[r] * 16 + c];
    else if (c < 32) v = attr[sidx[r] * 16 + (c - 16)];
    else if (c < 44) v = state[ridx[r] * 12 + (c - 32)];
    else if (c < 56) v = state[sidx[r] * 12 + (c - 44)];
    else if (c < 60) v = Ra[(size_t)(e0 + r) * 4 + (c - 56)];
    X[idx] = f2bf(v);
  }
  __syncthreads();
  const int lane = tid & 31, n0 = (tid >> 5) * 16;
  v8f a1 = {};
#pragma unroll
  for (int kc = 0; kc < 64; kc += 32)
    a1 = wmma_bf16(load_a_tile(X, 64, kc, lane), load_b_w(w1t, 64, kc, n0, lane), a1);
  epilogue_relu_bf16(H1, NF, n0, a1, b1, lane);
  __syncthreads();
  v8f a2 = {};
#pragma unroll
  for (int kc = 0; kc < NF; kc += 32)
    a2 = wmma_bf16(load_a_tile(H1, NF, kc, lane), load_b_w(w2t, NF, kc, n0, lane), a2);
  epilogue_relu_bf16(H2, NF, n0, a2, b2, lane);
  __syncthreads();
  v8f a3 = {};
#pragma unroll
  for (int kc = 0; kc < NF; kc += 32)
    a3 = wmma_bf16(load_a_tile(H2, NF, kc, lane), load_b_w(w3t, NF, kc, n0, lane), a3);
  epilogue_relu_bf16(ee + (size_t)e0 * NF, NF, n0, a3, b3, lane);
}

// ---- relation propagator ----------------------------------------------------
// A-tile (16 x 384 bf16 = 12KB) staged entirely with async global->LDS B128:
// per thread exactly 3 ops (ee row chunk, effect[recv] chunk, effect[send] chunk).
__global__ void __launch_bounds__(TPB) rel_prop_kernel(
    const uint16_t* __restrict__ ee, const uint16_t* __restrict__ effect,
    const int* __restrict__ recv, const int* __restrict__ send,
    const uint16_t* __restrict__ rp_wt, const float* __restrict__ rp_b,
    float* __restrict__ agg) {
  __shared__ __align__(16) uint16_t X[16 * 384];
  __shared__ int ridx[16], sidx[16];
  const int tid = threadIdx.x;
  const int e0 = blockIdx.x * 16;
  if (tid < 16)      ridx[tid] = recv[e0 + tid];
  else if (tid < 32) sidx[tid - 16] = send[e0 + tid - 16];
  __syncthreads();
  {
    const int r = tid >> 4;             // tile row 0..15
    const int cb = (tid & 15) * 16;     // byte chunk 0..240 within a 256B row
    char* xrow = (char*)X + r * 768;    // 384 bf16 = 768B per X row
    async_copy_b128(xrow + cb,       (const char*)(ee + (size_t)(e0 + r) * NF) + cb);
    async_copy_b128(xrow + 256 + cb, (const char*)(effect + (size_t)ridx[r] * NF) + cb);
    async_copy_b128(xrow + 512 + cb, (const char*)(effect + (size_t)sidx[r] * NF) + cb);
  }
  async_wait0();
  __syncthreads();
  const int lane = tid & 31, n0 = (tid >> 5) * 16;
  v8f acc = {};
#pragma unroll
  for (int kc = 0; kc < 384; kc += 32)
    acc = wmma_bf16(load_a_tile(X, 384, kc, lane), load_b_w(rp_wt, 384, kc, n0, lane), acc);
  const int col = lane & 15;
  const float bb = rp_b[n0 + col];
  const int rbase = (lane & 16) ? 8 : 0;
#pragma unroll
  for (int i = 0; i < 8; ++i) {
    float x = fmaxf(acc[i] + bb, 0.f);
    atomicAdd(&agg[(size_t)ridx[rbase + i] * NF + n0 + col], x);
  }
}

// ---- particle propagator: effect = relu([pe, agg] @ pp_w + b) ---------------
__global__ void __launch_bounds__(TPB) node_prop_kernel(
    const uint16_t* __restrict__ pe, const float* __restrict__ agg,
    const uint16_t* __restrict__ pp_wt, const float* __restrict__ pp_b,
    uint16_t* __restrict__ effect_out) {
  __shared__ __align__(16) uint16_t X[16 * 256];
  const int tid = threadIdx.x;
  const int row0 = blockIdx.x * 16;
  {
    // pe half: direct bf16 rows -> async copy (1 B128 per thread)
    const int r = tid >> 4;
    const int cb = (tid & 15) * 16;
    async_copy_b128((char*)X + r * 512 + cb,
                    (const char*)(pe + (size_t)(row0 + r) * NF) + cb);
  }
  // agg half: f32 -> bf16 conversion in VALU
  for (int idx = tid; idx < 16 * 128; idx += TPB) {
    int r = idx >> 7, c = idx & 127;
    X[r * 256 + 128 + c] = f2bf(agg[(size_t)(row0 + r) * NF + c]);
  }
  async_wait0();
  __syncthreads();
  const int lane = tid & 31, n0 = (tid >> 5) * 16;
  v8f acc = {};
#pragma unroll
  for (int kc = 0; kc < 256; kc += 32)
    acc = wmma_bf16(load_a_tile(X, 256, kc, lane), load_b_w(pp_wt, 256, kc, n0, lane), acc);
  epilogue_relu_bf16(effect_out + (size_t)row0 * NF, NF, n0, acc, pp_b, lane);
}

// ---- decoder: two 128x128 relu layers (WMMA) + 128->3 tail (VALU) -----------
__global__ void __launch_bounds__(TPB) decoder_kernel(
    const uint16_t* __restrict__ effect,
    const uint16_t* __restrict__ w1t, const float* __restrict__ b1,
    const uint16_t* __restrict__ w2t, const float* __restrict__ b2,
    const float* __restrict__ w3, const float* __restrict__ b3,
    float* __restrict__ out) {
  __shared__ __align__(16) uint16_t H1[16 * NF];
  __shared__ __align__(16) uint16_t H2[16 * NF];
  const int tid = threadIdx.x;
  const int row0 = blockIdx.x * 16;
  const int lane = tid & 31, n0 = (tid >> 5) * 16;
  const uint16_t* A0 = effect + (size_t)row0 * NF;  // A direct from global (row-major)
  v8f a1 = {};
#pragma unroll
  for (int kc = 0; kc < NF; kc += 32)
    a1 = wmma_bf16(load_a_tile(A0, NF, kc, lane), load_b_w(w1t, NF, kc, n0, lane), a1);
  epilogue_relu_bf16(H1, NF, n0, a1, b1, lane);
  __syncthreads();
  v8f a2 = {};
#pragma unroll
  for (int kc = 0; kc < NF; kc += 32)
    a2 = wmma_bf16(load_a_tile(H1, NF, kc, lane), load_b_w(w2t, NF, kc, n0, lane), a2);
  epilogue_relu_bf16(H2, NF, n0, a2, b2, lane);
  __syncthreads();
  if (tid < 48) {
    int r = tid / 3, j = tid % 3;
    float s = b3[j];
    for (int k = 0; k < NF; ++k) s += bf2f(H2[r * NF + k]) * w3[k * 3 + j];
    out[(size_t)(row0 + r) * 3 + j] = s;
  }
}

// ---- host launch ------------------------------------------------------------
extern "C" void kernel_launch(void* const* d_in, const int* in_sizes, int n_in,
                              void* d_out, int out_size, void* d_ws, size_t ws_size,
                              hipStream_t stream) {
  (void)in_sizes; (void)n_in; (void)out_size; (void)ws_size;
  const float* attr  = (const float*)d_in[0];
  const float* state = (const float*)d_in[1];
  const float* Ra    = (const float*)d_in[2];
  const int*   recv  = (const int*)d_in[3];
  const int*   send  = (const int*)d_in[4];
  // d_in[5] = pstep (constant 2 in reference; hardcoded below)
  const float* pe_w1 = (const float*)d_in[6];  const float* pe_b1 = (const float*)d_in[7];
  const float* pe_w2 = (const float*)d_in[8];  const float* pe_b2 = (const float*)d_in[9];
  const float* ee_w1 = (const float*)d_in[10]; const float* ee_b1 = (const float*)d_in[11];
  const float* ee_w2 = (const float*)d_in[12]; const float* ee_b2 = (const float*)d_in[13];
  const float* ee_w3 = (const float*)d_in[14]; const float* ee_b3 = (const float*)d_in[15];
  const float* rp_w  = (const float*)d_in[16]; const float* rp_b  = (const float*)d_in[17];
  const float* pp_w  = (const float*)d_in[18]; const float* pp_b  = (const float*)d_in[19];
  const float* de_w1 = (const float*)d_in[20]; const float* de_b1 = (const float*)d_in[21];
  const float* de_w2 = (const float*)d_in[22]; const float* de_b2 = (const float*)d_in[23];
  const float* de_w3 = (const float*)d_in[24]; const float* de_b3 = (const float*)d_in[25];

  char* ws = (char*)d_ws;
  size_t off = 0;
  auto take = [&](size_t bytes) -> char* {
    char* p = ws + off;
    off = (off + bytes + 255) & ~(size_t)255;
    return p;
  };
  uint16_t* pe_w1t = (uint16_t*)take((size_t)32 * NF * 2);
  uint16_t* pe_w2t = (uint16_t*)take((size_t)NF * NF * 2);
  uint16_t* ee_w1t = (uint16_t*)take((size_t)64 * NF * 2);
  uint16_t* ee_w2t = (uint16_t*)take((size_t)NF * NF * 2);
  uint16_t* ee_w3t = (uint16_t*)take((size_t)NF * NF * 2);
  uint16_t* rp_wt  = (uint16_t*)take((size_t)384 * NF * 2);
  uint16_t* pp_wt  = (uint16_t*)take((size_t)256 * NF * 2);
  uint16_t* de_w1t = (uint16_t*)take((size_t)NF * NF * 2);
  uint16_t* de_w2t = (uint16_t*)take((size_t)NF * NF * 2);
  uint16_t* peb  = (uint16_t*)take((size_t)NNODE * NF * 2);
  uint16_t* effb = (uint16_t*)take((size_t)NNODE * NF * 2);
  uint16_t* eeb  = (uint16_t*)take((size_t)NEDGE * NF * 2);
  float*    agg  = (float*)take((size_t)NNODE * NF * 4);

  auto conv = [&](const float* W, uint16_t* Wt, int Kin, int Kpad) {
    int total = NF * Kpad;
    conv_w_kernel<<<(total + 255) / 256, 256, 0, stream>>>(W, Wt, Kin, Kpad);
  };
  conv(pe_w1, pe_w1t, 28, 32);
  conv(pe_w2, pe_w2t, 128, 128);
  conv(ee_w1, ee_w1t, 60, 64);
  conv(ee_w2, ee_w2t, 128, 128);
  conv(ee_w3, ee_w3t, 128, 128);
  conv(rp_w,  rp_wt, 384, 384);
  conv(pp_w,  pp_wt, 256, 256);
  conv(de_w1, de_w1t, 128, 128);
  conv(de_w2, de_w2t, 128, 128);

  node_enc_kernel<<<NNODE / 16, TPB, 0, stream>>>(attr, state, pe_w1t, pe_b1, pe_w2t, pe_b2, peb);
  edge_enc_kernel<<<NEDGE / 16, TPB, 0, stream>>>(attr, state, Ra, recv, send,
                                                  ee_w1t, ee_b1, ee_w2t, ee_b2, ee_w3t, ee_b3, eeb);
  hipMemsetAsync(effb, 0, (size_t)NNODE * NF * 2, stream);  // effect starts at zero
  for (int step = 0; step < 2; ++step) {                    // pstep = 2
    hipMemsetAsync(agg, 0, (size_t)NNODE * NF * 4, stream);
    rel_prop_kernel<<<NEDGE / 16, TPB, 0, stream>>>(eeb, effb, recv, send, rp_wt, rp_b, agg);
    node_prop_kernel<<<NNODE / 16, TPB, 0, stream>>>(peb, agg, pp_wt, pp_b, effb);
  }
  decoder_kernel<<<NNODE / 16, TPB, 0, stream>>>(effb, de_w1t, de_b1, de_w2t, de_b2,
                                                 de_w3, de_b3, (float*)d_out);
}
